// Extractor_61125974557509
// MI455X (gfx1250) — matrix operand order repaired and checked by reading, not compiled, not run on hardware
//
#include <hip/hip_runtime.h>
#include <hip/hip_bf16.h>

typedef __attribute__((ext_vector_type(16))) __bf16      v16bf;
typedef __attribute__((ext_vector_type(8)))  float       v8f;
typedef __attribute__((ext_vector_type(8)))  unsigned int v8u;

#define N_NODES  100000
#define FEAT     128
#define NTILES   (N_NODES / 16)
#define BN_EPS   1e-5f

// ---------------- utility kernels ----------------

__global__ void k_zero(float* __restrict__ p, int n) {
  int i = blockIdx.x * blockDim.x + threadIdx.x;
  if (i < n) p[i] = 0.0f;
}

__global__ void k_degree(const int* __restrict__ dst, float* __restrict__ deg, int nE) {
  int e = blockIdx.x * blockDim.x + threadIdx.x;
  if (e < nE) atomicAdd(&deg[dst[e]], 1.0f);
}

// Wt[o][k] = W[o][k], converted to bf16 (B-fragment friendly: row o contiguous in k)
__global__ void k_wt(const float* __restrict__ W, __bf16* __restrict__ Wt) {
  int i = blockIdx.x * blockDim.x + threadIdx.x;  // 0..16383
  int k = i >> 7, o = i & 127;
  Wt[o * 128 + k] = (__bf16)W[o * 128 + k];
}

// scatter-add: agg[dst] += h[src], 1 edge per 32 threads, 4 floats/thread
__global__ void k_scatter(const float* __restrict__ h, const int* __restrict__ src,
                          const int* __restrict__ dst, float* __restrict__ agg, int nE) {
  int g = blockIdx.x * blockDim.x + threadIdx.x;
  int e = g >> 5, l = g & 31;
  if (e < nE) {
    int s = src[e], d = dst[e];
    const float4 v = *(const float4*)(h + (size_t)s * FEAT + l * 4);
    float* o = agg + (size_t)d * FEAT + l * 4;
    atomicAdd(o + 0, v.x);
    atomicAdd(o + 1, v.y);
    atomicAdd(o + 2, v.z);
    atomicAdd(o + 3, v.w);
  }
}

// ---------------- fused SAGE GEMM: out = L2norm(relu-pre)(agg/deg @ Wl^T + h @ Wr^T),
// then relu, store pre-BN, and accumulate per-feature sum/sumsq into stats ----------------

__global__ void __launch_bounds__(256) k_gemm(
    const float* __restrict__ h, const float* __restrict__ agg,
    const float* __restrict__ deg,
    const __bf16* __restrict__ Wlt, const __bf16* __restrict__ Wrt,
    float* __restrict__ out, float* __restrict__ stats) {
  __shared__ __attribute__((aligned(32))) __bf16 sW[2][128 * 128];
  __shared__ float sSum[128], sSq[128];

  {  // stage both weight matrices (bf16, k-major) into LDS
    const unsigned int* wl = (const unsigned int*)Wlt;
    const unsigned int* wr = (const unsigned int*)Wrt;
    unsigned int* dl = (unsigned int*)sW[0];
    unsigned int* dr = (unsigned int*)sW[1];
    for (int i = threadIdx.x; i < 128 * 128 / 2; i += 256) { dl[i] = wl[i]; dr[i] = wr[i]; }
    if (threadIdx.x < 128) { sSum[threadIdx.x] = 0.0f; sSq[threadIdx.x] = 0.0f; }
  }
  __syncthreads();

  const int lane = threadIdx.x & 31;
  const int wid  = threadIdx.x >> 5;
  const int tile = blockIdx.x * 8 + wid;

  if (tile < NTILES) {  // wave-uniform: EXEC stays all-ones inside (WMMA requirement)
    const int hi  = lane >> 4;           // half-wave select
    const int row = tile * 16 + (lane & 15);
    const float rdeg = 1.0f / fmaxf(deg[row], 1.0f);
    const float* ph = h   + (size_t)row * FEAT;
    const float* pa = agg + (size_t)row * FEAT;

    v8f acc[8] = {};

#pragma unroll
    for (int kb = 0; kb < 128; kb += 32) {
      // 16-bit A 16x32 layout: lanes<16 take K {kb..kb+7, kb+16..kb+23},
      // lanes>=16 take K {kb+8..kb+15, kb+24..kb+31}
      const int k0 = kb + hi * 8;
      float4 h0 = *(const float4*)(ph + k0);
      float4 h1 = *(const float4*)(ph + k0 + 4);
      float4 h2 = *(const float4*)(ph + k0 + 16);
      float4 h3 = *(const float4*)(ph + k0 + 20);
      float4 a0 = *(const float4*)(pa + k0);
      float4 a1 = *(const float4*)(pa + k0 + 4);
      float4 a2 = *(const float4*)(pa + k0 + 16);
      float4 a3 = *(const float4*)(pa + k0 + 20);

      v16bf aH, aA;
      aH[0]  = (__bf16)h0.x; aH[1]  = (__bf16)h0.y; aH[2]  = (__bf16)h0.z; aH[3]  = (__bf16)h0.w;
      aH[4]  = (__bf16)h1.x; aH[5]  = (__bf16)h1.y; aH[6]  = (__bf16)h1.z; aH[7]  = (__bf16)h1.w;
      aH[8]  = (__bf16)h2.x; aH[9]  = (__bf16)h2.y; aH[10] = (__bf16)h2.z; aH[11] = (__bf16)h2.w;
      aH[12] = (__bf16)h3.x; aH[13] = (__bf16)h3.y; aH[14] = (__bf16)h3.z; aH[15] = (__bf16)h3.w;
      aA[0]  = (__bf16)(a0.x * rdeg); aA[1]  = (__bf16)(a0.y * rdeg);
      aA[2]  = (__bf16)(a0.z * rdeg); aA[3]  = (__bf16)(a0.w * rdeg);
      aA[4]  = (__bf16)(a1.x * rdeg); aA[5]  = (__bf16)(a1.y * rdeg);
      aA[6]  = (__bf16)(a1.z * rdeg); aA[7]  = (__bf16)(a1.w * rdeg);
      aA[8]  = (__bf16)(a2.x * rdeg); aA[9]  = (__bf16)(a2.y * rdeg);
      aA[10] = (__bf16)(a2.z * rdeg); aA[11] = (__bf16)(a2.w * rdeg);
      aA[12] = (__bf16)(a3.x * rdeg); aA[13] = (__bf16)(a3.y * rdeg);
      aA[14] = (__bf16)(a3.z * rdeg); aA[15] = (__bf16)(a3.w * rdeg);

      const int bbase = (lane & 15) * 128 + kb + hi * 8;  // B col n = 16f+lane%16, K {kb+8h..+7, kb+16+8h..+7}
#pragma unroll
      for (int f = 0; f < 8; ++f) {
        const uint4 l0 = *(const uint4*)(&sW[0][bbase + 2048 * f]);
        const uint4 l1 = *(const uint4*)(&sW[0][bbase + 2048 * f + 16]);
        const uint4 r0 = *(const uint4*)(&sW[1][bbase + 2048 * f]);
        const uint4 r1 = *(const uint4*)(&sW[1][bbase + 2048 * f + 16]);
        v16bf bl = __builtin_bit_cast(v16bf, (v8u){l0.x, l0.y, l0.z, l0.w, l1.x, l1.y, l1.z, l1.w});
        v16bf br = __builtin_bit_cast(v16bf, (v8u){r0.x, r0.y, r0.z, r0.w, r1.x, r1.y, r1.z, r1.w});
        acc[f] = __builtin_amdgcn_wmma_f32_16x16x32_bf16(false, aA, false, bl,
                                                         (short)0, acc[f], false, false);
        acc[f] = __builtin_amdgcn_wmma_f32_16x16x32_bf16(false, aH, false, br,
                                                         (short)0, acc[f], false, false);
      }
    }

    // per-node L2 norm: C/D layout -> VGPR j is node m=j+8*hi, N = 16f + lane%16
    float scl[8];
#pragma unroll
    for (int j = 0; j < 8; ++j) {
      float s = 0.0f;
#pragma unroll
      for (int f = 0; f < 8; ++f) s += acc[f][j] * acc[f][j];
      s += __shfl_xor(s, 1, 32);
      s += __shfl_xor(s, 2, 32);
      s += __shfl_xor(s, 4, 32);
      s += __shfl_xor(s, 8, 32);
      scl[j] = 1.0f / fmaxf(sqrtf(s), 1e-12f);
    }

    // normalize + relu + store pre-BN + per-feature stats
#pragma unroll
    for (int f = 0; f < 8; ++f) {
      const int n = 16 * f + (lane & 15);
      float s1 = 0.0f, s2 = 0.0f;
#pragma unroll
      for (int j = 0; j < 8; ++j) {
        float v = fmaxf(acc[f][j] * scl[j], 0.0f);
        const int r = tile * 16 + j + 8 * hi;
        out[(size_t)r * FEAT + n] = v;
        s1 += v;
        s2 += v * v;
      }
      atomicAdd(&sSum[n], s1);
      atomicAdd(&sSq[n], s2);
    }
  }
  __syncthreads();
  if (threadIdx.x < 128) {
    atomicAdd(&stats[threadIdx.x], sSum[threadIdx.x]);
    atomicAdd(&stats[128 + threadIdx.x], sSq[threadIdx.x]);
  }
}

// training-mode BatchNorm applied in place (biased variance), float4 per thread
__global__ void k_bn(float* __restrict__ x, const float* __restrict__ stats,
                     const float* __restrict__ gamma, const float* __restrict__ beta, int n4) {
  int i = blockIdx.x * blockDim.x + threadIdx.x;
  if (i >= n4) return;
  const int f4 = (i & 31) * 4;
  float4 v = ((float4*)x)[i];
  float in[4] = {v.x, v.y, v.z, v.w};
  const float invN = 1.0f / (float)N_NODES;
#pragma unroll
  for (int t = 0; t < 4; ++t) {
    int f = f4 + t;
    float mean = stats[f] * invN;
    float var  = stats[128 + f] * invN - mean * mean;
    float sc   = gamma[f] * rsqrtf(var + BN_EPS);
    in[t] = sc * (in[t] - mean) + beta[f];
  }
  v.x = in[0]; v.y = in[1]; v.z = in[2]; v.w = in[3];
  ((float4*)x)[i] = v;
}

// ---------------- host orchestration ----------------

extern "C" void kernel_launch(void* const* d_in, const int* in_sizes, int n_in,
                              void* d_out, int out_size, void* d_ws, size_t ws_size,
                              hipStream_t stream) {
  const float* x   = (const float*)d_in[0];
  const int*   eic = (const int*)d_in[1];
  const int*   eid = (const int*)d_in[2];
  const float* W[8];
  for (int i = 0; i < 8; ++i) W[i] = (const float*)d_in[3 + i];   // conv1..4 {Wl,Wr}
  const float* bnp[8];
  for (int i = 0; i < 8; ++i) bnp[i] = (const float*)d_in[11 + i]; // bn1..4 {gamma,beta}

  const int E_C = 1600000, E_D = 200000;
  const int* srcC = eic;            const int* dstC = eic + E_C;
  const int* srcD = eid;            const int* dstD = eid + E_D;

  // workspace layout (agg and stats contiguous so one zero kernel covers both)
  char* p = (char*)d_ws;
  float* agg   = (float*)p; p += (size_t)N_NODES * FEAT * 4;
  float* stats = (float*)p; p += 256 * 4;
  float* hbuf  = (float*)p; p += (size_t)N_NODES * FEAT * 4;
  float* degc  = (float*)p; p += (size_t)N_NODES * 4;
  float* degd  = (float*)p; p += (size_t)N_NODES * 4;
  __bf16* wt[8];
  for (int i = 0; i < 8; ++i) { wt[i] = (__bf16*)p; p += 128 * 128 * 2; }

  float* hout = (float*)d_out;

  // degrees (once per launch)
  k_zero<<<(N_NODES + 255) / 256, 256, 0, stream>>>(degc, N_NODES);
  k_zero<<<(N_NODES + 255) / 256, 256, 0, stream>>>(degd, N_NODES);
  k_degree<<<(E_C + 255) / 256, 256, 0, stream>>>(dstC, degc, E_C);
  k_degree<<<(E_D + 255) / 256, 256, 0, stream>>>(dstD, degd, E_D);
  // bf16 transposed weights (once per launch)
  for (int i = 0; i < 8; ++i) k_wt<<<64, 256, 0, stream>>>(W[i], wt[i]);

  auto block = [&](const float* cur, float* nxt, const int* src, const int* dst, int nE,
                   const float* deg, const __bf16* wlt, const __bf16* wrt,
                   const float* g, const float* b) {
    k_zero<<<(N_NODES * FEAT + 256 + 255) / 256, 256, 0, stream>>>(agg, N_NODES * FEAT + 256);
    k_scatter<<<(nE * 32 + 255) / 256, 256, 0, stream>>>(cur, src, dst, agg, nE);
    k_gemm<<<(NTILES + 7) / 8, 256, 0, stream>>>(cur, agg, deg, wlt, wrt, nxt, stats);
    k_bn<<<(N_NODES * FEAT / 4 + 255) / 256, 256, 0, stream>>>(nxt, stats, g, b,
                                                               N_NODES * FEAT / 4);
  };

  // b1: conv1/bn1, connections         : x    -> d_out
  block(x,    hout, srcC, dstC, E_C, degc, wt[0], wt[1], bnp[0], bnp[1]);
  // b2-b4: conv4/bn2, connections (ping-pong)
  block(hout, hbuf, srcC, dstC, E_C, degc, wt[6], wt[7], bnp[2], bnp[3]);
  block(hbuf, hout, srcC, dstC, E_C, degc, wt[6], wt[7], bnp[2], bnp[3]);
  block(hout, hbuf, srcC, dstC, E_C, degc, wt[6], wt[7], bnp[2], bnp[3]);
  // b5: conv2/bn3, destinations        : hbuf -> d_out
  block(hbuf, hout, srcD, dstD, E_D, degd, wt[2], wt[3], bnp[4], bnp[5]);
  // b6-b7: conv3/bn4, connections      : ... -> d_out (final lands in d_out)
  block(hout, hbuf, srcC, dstC, E_C, degc, wt[4], wt[5], bnp[6], bnp[7]);
  block(hbuf, hout, srcC, dstC, E_C, degc, wt[4], wt[5], bnp[6], bnp[7]);
}
